// MambaBlock_48223892799527
// MI455X (gfx1250) — matrix operand order repaired and checked by reading, not compiled, not run on hardware
//
#include <hip/hip_runtime.h>
#include <hip/hip_bf16.h>

// ---------------------------------------------------------------------------
// Mamba block for MI455X (gfx1250, wave32, WMMA).
// GEMMs: v_wmma_f32_16x16x32_f16, LDS-staged 64x64 tiles, branchless staging
// (weights pre-transposed+padded), async global->LDS staging (ASYNCcnt).
// ---------------------------------------------------------------------------

typedef __attribute__((ext_vector_type(16))) _Float16 v16h;
typedef __attribute__((ext_vector_type(8)))  float    v8f;

union Frag16 {
    v16h  h;
    uint4 q[2];
};

#define D_MODEL  1024
#define D_INNER  2048
#define D_STATE  16
#define DT_RANK  64
#define NROWS    4096        // B*L = 2*2048
#define SEQ_L    2048
#define NBATCH   2
#define DBL_LD   128         // padded leading dim of dbl (96 -> 128)

#define TILE_M 64
#define TILE_N 64
#define TILE_K 32

// ---- gfx1250 async global->LDS staging ------------------------------------
#if defined(__has_builtin)
# if __has_builtin(__builtin_amdgcn_global_load_async_to_lds_b128) && \
     __has_builtin(__builtin_amdgcn_s_wait_asynccnt)
#  define ASYNC_LDS 1
# endif
#endif

// exact pointee type from the compiler diagnostic:
//   __attribute__((__vector_size__(4*sizeof(int)))) int __device__ *
typedef int gcc_int4 __attribute__((vector_size(16)));
typedef __attribute__((address_space(1))) gcc_int4* as1_int4_p;
typedef __attribute__((address_space(3))) gcc_int4* as3_int4_p;

__device__ __forceinline__ void copy_b128_g2l(const _Float16* g, _Float16* l) {
#if defined(ASYNC_LDS)
    __builtin_amdgcn_global_load_async_to_lds_b128(
        (as1_int4_p)(void*)g, (as3_int4_p)(void*)l, 0, 0);
#else
    *(uint4*)l = *(const uint4*)g;
#endif
}

__device__ __forceinline__ void staging_wait() {
#if defined(ASYNC_LDS)
    __builtin_amdgcn_s_wait_asynccnt(0);
#endif
}

// ---------------- WMMA GEMM:  C[M,N] = A[M,K] * Bt[N,K]^T  (f16 in, f32 out)
// A:  [M,K] row-major f16.   Bt: transposed weights, [N,K] row-major f16.
// Requirements: M % 64 == 0, N % 64 == 0, K % 32 == 0 (caller pads).
__device__ __forceinline__ void load_a_frag(Frag16& f, const _Float16* As,
                                            int rowoff, int lane) {
    const int m    = rowoff + (lane & 15);
    const int base = (lane >> 4) * 8;              // 0 or 8
    f.q[0] = *(const uint4*)(As + m * TILE_K + base);      // K = base .. base+7
    f.q[1] = *(const uint4*)(As + m * TILE_K + base + 16); // K = base+16 .. base+23
}

__device__ __forceinline__ void load_b_frag(Frag16& f, const _Float16* Bs,
                                            int coloff, int lane) {
    const int n    = coloff + (lane & 15);
    const int base = (lane >> 4) * 8;              // 0 or 8
    const uint4* p = (const uint4*)(Bs + n * TILE_K + base);
    f.q[0] = p[0];                                 // K = base .. base+7
    f.q[1] = p[2];                                 // K = base+16 .. base+23
}

__global__ __launch_bounds__(128)
void wmma_gemm_f16f32(const _Float16* __restrict__ A,
                      const _Float16* __restrict__ Bt,
                      float* __restrict__ C,
                      int M, int N, int K) {
    __shared__ __align__(16) _Float16 As[TILE_M * TILE_K]; // [m][k]
    __shared__ __align__(16) _Float16 Bs[TILE_N * TILE_K]; // [n][k]

    const int tid  = threadIdx.x;
    const int lane = tid & 31;
    const int wave = tid >> 5;             // 0..3
    const int wm   = (wave >> 1) * 32;     // wave row offset in block tile
    const int wn   = (wave & 1)  * 32;     // wave col offset in block tile
    const int row0 = blockIdx.y * TILE_M;
    const int col0 = blockIdx.x * TILE_N;

    const v8f vzero = {0.f, 0.f, 0.f, 0.f, 0.f, 0.f, 0.f, 0.f};
    v8f acc[2][2];
    acc[0][0] = vzero; acc[0][1] = vzero; acc[1][0] = vzero; acc[1][1] = vzero;

    // staging assignment: 128 threads x 16 halves = one 64x32 tile
    const int sr = tid >> 1;               // 0..63  tile row
    const int sk = (tid & 1) * 16;         // K half: 0 or 16

    const _Float16* ga = A  + (size_t)(row0 + sr) * K + sk;
    const _Float16* gb = Bt + (size_t)(col0 + sr) * K + sk;
    _Float16* la = As + sr * TILE_K + sk;
    _Float16* lb = Bs + sr * TILE_K + sk;

    for (int k0 = 0; k0 < K; k0 += TILE_K) {
        // ---- stage A and Bt tiles (64 x 32 each), pure b128 traffic -------
        copy_b128_g2l(ga + k0,     la);
        copy_b128_g2l(ga + k0 + 8, la + 8);
        copy_b128_g2l(gb + k0,     lb);
        copy_b128_g2l(gb + k0 + 8, lb + 8);
        if (k0 + TILE_K < K) {
            __builtin_prefetch(ga + k0 + TILE_K, 0, 1);   // global_prefetch_b8
            __builtin_prefetch(gb + k0 + TILE_K, 0, 1);
        }
        staging_wait();
        __syncthreads();

        Frag16 a0, a1, b0, b1;
        load_a_frag(a0, As, wm,      lane);
        load_a_frag(a1, As, wm + 16, lane);
        load_b_frag(b0, Bs, wn,      lane);
        load_b_frag(b1, Bs, wn + 16, lane);

        acc[0][0] = __builtin_amdgcn_wmma_f32_16x16x32_f16(
            false, a0.h, false, b0.h, (short)0, acc[0][0], false, false);
        acc[0][1] = __builtin_amdgcn_wmma_f32_16x16x32_f16(
            false, a0.h, false, b1.h, (short)0, acc[0][1], false, false);
        acc[1][0] = __builtin_amdgcn_wmma_f32_16x16x32_f16(
            false, a1.h, false, b0.h, (short)0, acc[1][0], false, false);
        acc[1][1] = __builtin_amdgcn_wmma_f32_16x16x32_f16(
            false, a1.h, false, b1.h, (short)0, acc[1][1], false, false);

        __syncthreads();
    }

    // ---- epilogue: C/D layout -> VGPR r holds M = r + 8*(lane>=16), N = lane&15
    const int rsel = (lane >> 4) * 8;
    const int csel = lane & 15;
    #pragma unroll
    for (int i = 0; i < 2; ++i) {
        #pragma unroll
        for (int j = 0; j < 2; ++j) {
            const int mbase = row0 + wm + i * 16 + rsel;
            const int nidx  = col0 + wn + j * 16 + csel;
            #pragma unroll
            for (int r = 0; r < 8; ++r) {
                C[(size_t)(mbase + r) * N + nidx] = acc[i][j][r];
            }
        }
    }
}

// ---------------- LayerNorm over D_MODEL, output f16 -----------------------
__global__ __launch_bounds__(256)
void layernorm_kernel(const float* __restrict__ x,
                      const float* __restrict__ gamma,
                      const float* __restrict__ beta,
                      _Float16* __restrict__ out) {
    __shared__ float red[256];
    const int row = blockIdx.x;
    const int tid = threadIdx.x;
    const float* xr = x + (size_t)row * D_MODEL;

    float s = 0.f;
    for (int i = tid; i < D_MODEL; i += 256) s += xr[i];
    red[tid] = s; __syncthreads();
    for (int o = 128; o > 0; o >>= 1) {
        if (tid < o) red[tid] += red[tid + o];
        __syncthreads();
    }
    const float mu = red[0] * (1.f / D_MODEL);
    __syncthreads();

    float v = 0.f;
    for (int i = tid; i < D_MODEL; i += 256) {
        const float d = xr[i] - mu;
        v += d * d;
    }
    red[tid] = v; __syncthreads();
    for (int o = 128; o > 0; o >>= 1) {
        if (tid < o) red[tid] += red[tid + o];
        __syncthreads();
    }
    const float rstd = rsqrtf(red[0] * (1.f / D_MODEL) + 1e-5f);

    for (int i = tid; i < D_MODEL; i += 256) {
        const float xn = (xr[i] - mu) * rstd * gamma[i] + beta[i];
        out[(size_t)row * D_MODEL + i] = (_Float16)xn;
    }
}

// ---------------- weight prep: f32 [K,N] -> f16 transposed [Npad,K] --------
__global__ void cvt_transpose_f16(const float* __restrict__ in,  // [K, N]
                                  _Float16* __restrict__ out,    // [Npad, K]
                                  int K, int N, int Npad) {
    const int i = blockIdx.x * blockDim.x + threadIdx.x;
    if (i >= Npad * K) return;
    const int n = i / K;
    const int k = i - n * K;
    out[i] = (n < N) ? (_Float16)in[(size_t)k * N + n] : (_Float16)0.f;
}

// ---------------- causal depthwise conv (k=4) + bias + SiLU ----------------
__global__ void conv_silu_kernel(const float* __restrict__ xz,  // [4096, 4096], u = cols 0..2047
                                 const float* __restrict__ cw,  // [2048, 4]
                                 const float* __restrict__ cb,  // [2048]
                                 float* __restrict__ u2,        // [4096, 2048] f32
                                 _Float16* __restrict__ u2h) {  // f16 copy for WMMA
    const int idx = blockIdx.x * blockDim.x + threadIdx.x;
    if (idx >= NROWS * D_INNER) return;
    const int row = idx >> 11;        // /2048
    const int d   = idx & (D_INNER - 1);
    const int b   = row >> 11;        // /2048 (L = 2048)
    const int l   = row & (SEQ_L - 1);

    float acc = cb[d];
    #pragma unroll
    for (int j = 0; j < 4; ++j) {
        const int ls = l - 3 + j;
        if (ls >= 0)
            acc += cw[d * 4 + j] * xz[(size_t)(b * SEQ_L + ls) * (2 * D_INNER) + d];
    }
    const float s = acc / (1.f + __expf(-acc));   // SiLU
    u2[idx]  = s;
    u2h[idx] = (_Float16)s;
}

// ---------------- extract dt_low (cols 0..63 of dbl, ld=128) as f16 --------
__global__ void dtlow_kernel(const float* __restrict__ dbl,
                             _Float16* __restrict__ out) {
    const int i = blockIdx.x * blockDim.x + threadIdx.x;
    if (i >= NROWS * DT_RANK) return;
    const int row = i >> 6;
    const int j   = i & (DT_RANK - 1);
    out[i] = (_Float16)dbl[(size_t)row * DBL_LD + j];
}

// ---------------- dt = softplus(dt_pre + b_dt), in place -------------------
__global__ void softplus_kernel(float* __restrict__ dt,
                                const float* __restrict__ b_dt) {
    const int i = blockIdx.x * blockDim.x + threadIdx.x;
    if (i >= NROWS * D_INNER) return;
    const int d = i & (D_INNER - 1);
    const float v = dt[i] + b_dt[d];
    dt[i] = (v > 20.f) ? v : log1pf(__expf(v));
}

// ---------------- A = -exp(A_log) ------------------------------------------
__global__ void aprep_kernel(const float* __restrict__ A_log,
                             float* __restrict__ Aneg) {
    const int i = blockIdx.x * blockDim.x + threadIdx.x;
    if (i < D_INNER * D_STATE) Aneg[i] = -__expf(A_log[i]);
}

// ---------------- selective scan: one thread per (batch, channel) ----------
__global__ __launch_bounds__(256)
void scan_kernel(const float* __restrict__ dt,    // [4096, 2048]
                 const float* __restrict__ u2,    // [4096, 2048]
                 const float* __restrict__ dbl,   // [4096, 128] (B = 64..79, C = 80..95)
                 const float* __restrict__ Aneg,  // [2048, 16]
                 float* __restrict__ y) {         // [4096, 2048]
    const int idx = blockIdx.x * blockDim.x + threadIdx.x;
    if (idx >= NBATCH * D_INNER) return;
    const int b = idx >> 11;
    const int d = idx & (D_INNER - 1);

    float a[D_STATE], h[D_STATE];
    #pragma unroll
    for (int n = 0; n < D_STATE; ++n) {
        a[n] = Aneg[d * D_STATE + n];
        h[n] = 0.f;
    }

    for (int l = 0; l < SEQ_L; ++l) {
        const int row = b * SEQ_L + l;
        const float dtv = dt[(size_t)row * D_INNER + d];
        const float uv  = u2[(size_t)row * D_INNER + d];
        const float du  = dtv * uv;
        const float* bc = dbl + (size_t)row * DBL_LD;
        float yv = 0.f;
        #pragma unroll
        for (int n = 0; n < D_STATE; ++n) {
            const float dA = __expf(dtv * a[n]);
            h[n] = dA * h[n] + du * bc[64 + n];
            yv  += h[n] * bc[80 + n];
        }
        y[(size_t)row * D_INNER + d] = yv;
    }
}

// ---------------- y = (yscan + u2*D) * silu(z), write f16 ------------------
__global__ void gate_kernel(const float* __restrict__ yscan,
                            const float* __restrict__ u2,
                            const float* __restrict__ D,
                            const float* __restrict__ xz,   // z = cols 2048..4095
                            _Float16* __restrict__ yh) {
    const int i = blockIdx.x * blockDim.x + threadIdx.x;
    if (i >= NROWS * D_INNER) return;
    const int row = i >> 11;
    const int d   = i & (D_INNER - 1);
    const float z = xz[(size_t)row * (2 * D_INNER) + D_INNER + d];
    const float y = (yscan[i] + u2[i] * D[d]) * (z / (1.f + __expf(-z)));
    yh[i] = (_Float16)y;
}

// ---------------- out = x + yout -------------------------------------------
__global__ void residual_kernel(const float* __restrict__ x,
                                const float* __restrict__ yout,
                                float* __restrict__ out) {
    const int i = blockIdx.x * blockDim.x + threadIdx.x;
    if (i < NROWS * D_MODEL) out[i] = x[i] + yout[i];
}

// ===========================================================================
extern "C" void kernel_launch(void* const* d_in, const int* in_sizes, int n_in,
                              void* d_out, int out_size, void* d_ws, size_t ws_size,
                              hipStream_t stream) {
    const float* x      = (const float*)d_in[0];
    const float* gamma  = (const float*)d_in[1];
    const float* beta   = (const float*)d_in[2];
    const float* W_in   = (const float*)d_in[3];
    const float* conv_w = (const float*)d_in[4];
    const float* conv_b = (const float*)d_in[5];
    const float* W_x    = (const float*)d_in[6];
    const float* W_dt   = (const float*)d_in[7];
    const float* b_dt   = (const float*)d_in[8];
    const float* A_log  = (const float*)d_in[9];
    const float* Dv     = (const float*)d_in[10];
    const float* W_out  = (const float*)d_in[11];
    float* out = (float*)d_out;

    // ---- workspace layout (regions reused across non-overlapping lifetimes)
    char* ws = (char*)d_ws;
    size_t off = 0;
    auto take = [&](size_t bytes) {
        char* p = ws + off;
        off += (bytes + 255) & ~(size_t)255;
        return p;
    };
    // f16 A-operand region: xn (8.4MB) -> u2h (16.8MB) -> yh (16.8MB), serial reuse
    _Float16* a16   = (_Float16*)take((size_t)NROWS * D_INNER * 2);
    // f16 transposed-weight region: W_in^T -> W_x^T -> W_dt^T -> W_out^T, serial reuse
    _Float16* wT16  = (_Float16*)take((size_t)D_MODEL * 2 * D_INNER * 2);
    _Float16* dtl16 = (_Float16*)take((size_t)NROWS * DT_RANK * 2);
    float* xz    = (float*)take((size_t)NROWS * 2 * D_INNER * 4);   // 67 MB
    float* u2f   = (float*)take((size_t)NROWS * D_INNER * 4);       // 33.5 MB
    float* dbl   = (float*)take((size_t)NROWS * DBL_LD * 4);        // 2 MB (padded)
    float* dtf   = (float*)take((size_t)NROWS * D_INNER * 4);       // 33.5 MB
    float* Aneg  = (float*)take((size_t)D_INNER * D_STATE * 4);
    float* yscan = (float*)take((size_t)NROWS * D_INNER * 4);       // 33.5 MB
    float* yout  = yscan;  // reuse: yscan dead after gate, yout only needs 16.8 MB
    (void)ws_size; (void)in_sizes; (void)n_in; (void)out_size;

    const int T = 256;

    // 1. layernorm -> a16 (xn, f16)
    layernorm_kernel<<<NROWS, T, 0, stream>>>(x, gamma, beta, a16);

    // 2. W_in^T f16 ; GEMM1: xz[4096,4096] = xn[4096,1024] @ W_in[1024,4096]
    {
        const int n = (2 * D_INNER) * D_MODEL;
        cvt_transpose_f16<<<(n + T - 1) / T, T, 0, stream>>>(
            W_in, wT16, D_MODEL, 2 * D_INNER, 2 * D_INNER);
        dim3 g(2 * D_INNER / TILE_N, NROWS / TILE_M);
        wmma_gemm_f16f32<<<g, 128, 0, stream>>>(a16, wT16, xz, NROWS, 2 * D_INNER, D_MODEL);
    }

    // 3. depthwise causal conv + SiLU -> u2f (f32) + a16 (f16)
    conv_silu_kernel<<<(NROWS * D_INNER + T - 1) / T, T, 0, stream>>>(
        xz, conv_w, conv_b, u2f, a16);

    // 4. W_x^T f16 (96 -> pad 128) ; GEMM2: dbl[4096,128] = u2 @ W_x
    {
        const int n = DBL_LD * D_INNER;
        cvt_transpose_f16<<<(n + T - 1) / T, T, 0, stream>>>(
            W_x, wT16, D_INNER, DT_RANK + 2 * D_STATE, DBL_LD);
        dim3 g(DBL_LD / TILE_N, NROWS / TILE_M);
        wmma_gemm_f16f32<<<g, 128, 0, stream>>>(a16, wT16, dbl, NROWS, DBL_LD, D_INNER);
    }

    // 5. dt_low f16 ; W_dt^T f16 ; GEMM: dtf[4096,2048] = dt_low[4096,64] @ W_dt
    {
        int n = NROWS * DT_RANK;
        dtlow_kernel<<<(n + T - 1) / T, T, 0, stream>>>(dbl, dtl16);
        n = D_INNER * DT_RANK;
        cvt_transpose_f16<<<(n + T - 1) / T, T, 0, stream>>>(
            W_dt, wT16, DT_RANK, D_INNER, D_INNER);
        dim3 g(D_INNER / TILE_N, NROWS / TILE_M);
        wmma_gemm_f16f32<<<g, 128, 0, stream>>>(dtl16, wT16, dtf, NROWS, D_INNER, DT_RANK);
    }

    // 6. dt = softplus(dt_pre + b_dt) ; A = -exp(A_log)
    softplus_kernel<<<(NROWS * D_INNER + T - 1) / T, T, 0, stream>>>(dtf, b_dt);
    aprep_kernel<<<(D_INNER * D_STATE + T - 1) / T, T, 0, stream>>>(A_log, Aneg);

    // 7. selective scan over L (parallel over batch x channel)
    scan_kernel<<<(NBATCH * D_INNER + T - 1) / T, T, 0, stream>>>(
        dtf, u2f, dbl, Aneg, yscan);

    // 8. gate: y = (yscan + u2*D) * silu(z)  -> a16 (f16)
    gate_kernel<<<(NROWS * D_INNER + T - 1) / T, T, 0, stream>>>(
        yscan, u2f, Dv, xz, a16);

    // 9. W_out^T f16 ; GEMM3: yout[4096,1024] = y[4096,2048] @ W_out[2048,1024]
    {
        const int n = D_MODEL * D_INNER;
        cvt_transpose_f16<<<(n + T - 1) / T, T, 0, stream>>>(
            W_out, wT16, D_INNER, D_MODEL, D_MODEL);
        dim3 g(D_MODEL / TILE_N, NROWS / TILE_M);
        wmma_gemm_f16f32<<<g, 128, 0, stream>>>(a16, wT16, yout, NROWS, D_MODEL, D_INNER);
    }

    // 10. residual: out = x + yout
    residual_kernel<<<(NROWS * D_MODEL + T - 1) / T, T, 0, stream>>>(x, yout, out);
}
